// ConvPoint_21345987461169
// MI455X (gfx1250) — matrix operand order repaired and checked by reading, not compiled, not run on hardware
//
#include <hip/hip_runtime.h>

typedef _Float16 v16h __attribute__((ext_vector_type(16)));
typedef _Float16 v8h  __attribute__((ext_vector_type(8)));
typedef float    v8f  __attribute__((ext_vector_type(8)));

#define B_DIM  8
#define M_TOT  8192
#define I_DIM  64
#define O_DIM  64
#define K_NBR  16
#define KS     16
#define KK     (I_DIM * KS)   /* 1024 contraction length */
#define TM     16             /* m-tile per block */
#define FSTR   1032           /* padded F row stride in halves: 516 dwords -> 4-bank skew, 16B aligned */

// ---------------- kernel 0: cv_w f32 -> f16 (layout identical: o*1024 + i*16 + s) ----------
__global__ void cvt_cv_kernel(const float* __restrict__ cv, _Float16* __restrict__ cvh, int n) {
    int t = blockIdx.x * blockDim.x + threadIdx.x;
    if (t < n) cvh[t] = (_Float16)cv[t];
}

// ---------------- fused ConvPoint kernel: 1 block = (b, 16 support points) ------------------
__global__ __launch_bounds__(128)
void convpoint_kernel(const float* __restrict__ x,        // (B, I, N)
                      const float* __restrict__ pos,      // (B, 3, N)
                      const float* __restrict__ sup,      // (B, 3, N)
                      const int*   __restrict__ indices,  // (B, N, K)
                      const float* __restrict__ centers,  // (3, 16)
                      const float* __restrict__ W1, const float* __restrict__ b1,
                      const float* __restrict__ W2, const float* __restrict__ b2,
                      const float* __restrict__ W3, const float* __restrict__ b3,
                      const _Float16* __restrict__ cvh,   // (O, 1024) f16
                      const float* __restrict__ cvb,      // (O,)
                      float* __restrict__ out)            // (B, O, N)
{
    __shared__ float wW1[16 * 48];
    __shared__ float wW2[256];
    __shared__ float wW3[256];
    __shared__ float wb1[16], wb2[16], wb3[16];
    __shared__ float wC[48];
    __shared__ int   idxsh[TM * K_NBR];
    __shared__ float sqsh[TM * K_NBR];
    __shared__ __attribute__((aligned(16))) _Float16 matsh[TM * K_NBR * KS]; // 8 KB
    __shared__ __attribute__((aligned(16))) _Float16 Fsh[TM * FSTR];         // ~33 KB

    const int tid   = threadIdx.x;
    const int b     = blockIdx.x >> 9;          // 512 tiles per batch
    const int mbase = (blockIdx.x & 511) * TM;

    // ---- cooperative parameter staging ----
    for (int t = tid; t < 16 * 48; t += 128) wW1[t] = W1[t];
    for (int t = tid; t < 256;     t += 128) { wW2[t] = W2[t]; wW3[t] = W3[t]; }
    if (tid < 16) { wb1[tid] = b1[tid]; wb2[tid] = b2[tid]; wb3[tid] = b3[tid]; }
    if (tid < 48) wC[tid] = centers[tid];
    {
        const size_t base = ((size_t)b * M_TOT + mbase) * K_NBR;
        for (int t = tid; t < TM * K_NBR; t += 128) idxsh[t] = indices[base + t];
    }
    __syncthreads();

    // ---- phase 0: geometry (pairs p = tid, tid+128 over 16 m x 16 k sites) ----
    float ptsr[2][3];
    #pragma unroll
    for (int pp = 0; pp < 2; ++pp) {
        const int p = tid + pp * 128;
        const int m = p >> 4;
        const int n = idxsh[p];
        float s2 = 0.f;
        #pragma unroll
        for (int d = 0; d < 3; ++d) {
            float v = pos[((size_t)b * 3 + d) * M_TOT + n]
                    - sup[((size_t)b * 3 + d) * M_TOT + mbase + m];
            ptsr[pp][d] = v;
            s2 += v * v;
        }
        sqsh[p] = s2;
    }
    __syncthreads();

    // ---- phase 0b: per-neighbor kernel-weight MLP -> matsh ----
    #pragma unroll
    for (int pp = 0; pp < 2; ++pp) {
        const int p = tid + pp * 128;
        const int m = p >> 4;
        float mx = 0.f;
        #pragma unroll
        for (int k2 = 0; k2 < 16; ++k2) mx = fmaxf(mx, sqsh[m * 16 + k2]);
        float maxi = sqrtf(mx);
        if (maxi == 0.f) maxi = 1.f;
        const float inv = 1.f / maxi;

        float rel[48];
        #pragma unroll
        for (int d = 0; d < 3; ++d) {
            const float pn = ptsr[pp][d] * inv;
            #pragma unroll
            for (int s = 0; s < 16; ++s) rel[d * 16 + s] = pn - wC[d * 16 + s];
        }
        float h1[16], h2[16];
        #pragma unroll
        for (int t = 0; t < 16; ++t) {
            float a = wb1[t];
            #pragma unroll
            for (int j = 0; j < 48; ++j) a = fmaf(wW1[t * 48 + j], rel[j], a);
            h1[t] = fmaxf(a, 0.f);
        }
        #pragma unroll
        for (int t = 0; t < 16; ++t) {
            float a = wb2[t];
            #pragma unroll
            for (int u = 0; u < 16; ++u) a = fmaf(wW2[t * 16 + u], h1[u], a);
            h2[t] = fmaxf(a, 0.f);
        }
        #pragma unroll
        for (int s = 0; s < 16; ++s) {
            float a = wb3[s];
            #pragma unroll
            for (int u = 0; u < 16; ++u) a = fmaf(wW3[s * 16 + u], h2[u], a);
            matsh[p * 16 + s] = (_Float16)fmaxf(a, 0.f);
        }
    }
    __syncthreads();

    // ---- phase 1: gather feats + F[(i,s), m] = sum_k feats[i,k] * mat[m,k,s] -> LDS (f16) ----
    for (int q = 0; q < 8; ++q) {
        const int p = tid + q * 128;        // 0..1023
        const int i = p >> 4;
        const int m = p & 15;
        float fe[16];
        #pragma unroll
        for (int k = 0; k < 16; ++k)
            fe[k] = x[((size_t)b * I_DIM + i) * M_TOT + idxsh[m * 16 + k]];
        #pragma unroll
        for (int s = 0; s < 16; ++s) {
            float a = 0.f;
            #pragma unroll
            for (int k = 0; k < 16; ++k)
                a = fmaf(fe[k], (float)matsh[(m * 16 + k) * 16 + s], a);
            Fsh[m * FSTR + i * 16 + s] = (_Float16)a;
        }
    }
    __syncthreads();

    // ---- phase 2: OUT(16x16 per wave) = CV(16x1024) @ F(1024x16) via v_wmma_f32_16x16x32_f16 ----
    const int w    = tid >> 5;     // wave id 0..3 -> o-chunk
    const int lane = tid & 31;
    const int col  = lane & 15;    // A row (M=o_local) and B column (N=m_local)
    const int hi   = lane >> 4;

    // A fragment per lane: row o, halves K = {0..7,16..23} (lanes<16) or {8..15,24..31}
    const _Float16* pA = cvh + (size_t)(w * 16 + col) * KK + hi * 8;
    // B fragment per lane: column m, halves K = {0..7,16..23} (lanes<16) or {8..15,24..31}
    const _Float16* pB = Fsh + col * FSTR + hi * 8;

    v8f c = {};
    #pragma unroll 4
    for (int kk0 = 0; kk0 < KK; kk0 += 32) {
        v8h alo = *(const v8h*)(pA + kk0);
        v8h ahi = *(const v8h*)(pA + kk0 + 16);
        v8h blo = *(const v8h*)(pB + kk0);
        v8h bhi = *(const v8h*)(pB + kk0 + 16);
        v16h a, bv;
        #pragma unroll
        for (int j = 0; j < 8; ++j) {
            a[j] = alo[j];  a[8 + j] = ahi[j];
            bv[j] = blo[j]; bv[8 + j] = bhi[j];
        }
        c = __builtin_amdgcn_wmma_f32_16x16x32_f16(false, a, false, bv,
                                                   (short)0, c, false, false);
    }

    // ---- store: C/D layout — vreg r, lane L: row = r + 8*(L>=16), col = L%16 ----
    #pragma unroll
    for (int r = 0; r < 8; ++r) {
        const int o = w * 16 + r + 8 * hi;
        out[((size_t)b * O_DIM + o) * M_TOT + mbase + col] = c[r] + cvb[o];
    }
}

extern "C" void kernel_launch(void* const* d_in, const int* in_sizes, int n_in,
                              void* d_out, int out_size, void* d_ws, size_t ws_size,
                              hipStream_t stream) {
    const float* x       = (const float*)d_in[0];
    const float* pos     = (const float*)d_in[1];
    const float* sup     = (const float*)d_in[2];
    const int*   indices = (const int*)d_in[3];
    const float* centers = (const float*)d_in[4];
    const float* W1      = (const float*)d_in[5];
    const float* b1      = (const float*)d_in[6];
    const float* W2      = (const float*)d_in[7];
    const float* b2      = (const float*)d_in[8];
    const float* W3      = (const float*)d_in[9];
    const float* b3      = (const float*)d_in[10];
    const float* cv_w    = (const float*)d_in[11];
    const float* cv_b    = (const float*)d_in[12];
    float*       out     = (float*)d_out;

    _Float16* cvh = (_Float16*)d_ws;   // 64*1024 halves = 128 KB, L2-resident

    cvt_cv_kernel<<<(O_DIM * KK + 255) / 256, 256, 0, stream>>>(cv_w, cvh, O_DIM * KK);

    const int tiles = B_DIM * (M_TOT / TM);   // 8 * 512 = 4096 blocks
    convpoint_kernel<<<tiles, 128, 0, stream>>>(x, pos, sup, indices, centers,
                                                W1, b1, W2, b2, W3, b3,
                                                cvh, cv_b, out);
}
